// ConvBlock_2886218022989
// MI455X (gfx1250) — matrix operand order repaired and checked by reading, not compiled, not run on hardware
//
#include <hip/hip_runtime.h>

// ---------------------------------------------------------------------------
// DGCNN EdgeConv block for MI455X (gfx1250, wave32, WMMA bf16)
// B=8, C=3, N=4096, K=40, F=64
// ---------------------------------------------------------------------------

typedef __attribute__((ext_vector_type(16))) __bf16 v16bf;
typedef __attribute__((ext_vector_type(8)))  float  v8f;

#define B_    8
#define N_    4096
#define K_    40
#define F_    64
#define CNT_  1310720      // B*N*K
#define EPS_  1e-5f
#define SLOPE_ 0.2f

#define IDX_ELEMS 1310720  // B*N*K ints at start of ws
// float accumulators after idx:
#define OFF_M1   0         // 6
#define OFF_M2   6         // 21 (sym 6x6)
#define OFF_M1A  27        // 64 (first moment of a1)
#define OFF_S    91        // 4096 (a1 a1^T second moment)
#define OFF_SC1  4187
#define OFF_SH1  4251
#define OFF_SC2  4315
#define OFF_SH2  4379
#define ACC_ZERO_N 4187    // zero M1..S each call

__device__ __forceinline__ int sym6(int c, int d) {
  if (c > d) { int t = c; c = d; d = t; }
  return c * 6 - c * (c - 1) / 2 + (d - c);
}

__device__ __forceinline__ void load_edge(const float* __restrict__ x,
                                          int b, int i, int j, float e[6]) {
  const float* xb = x + b * 3 * N_;
  float xi0 = xb[i], xi1 = xb[N_ + i], xi2 = xb[2 * N_ + i];
  float xj0 = xb[j], xj1 = xb[N_ + j], xj2 = xb[2 * N_ + j];
  e[0] = xj0 - xi0; e[1] = xj1 - xi1; e[2] = xj2 - xi2;
  e[3] = xi0; e[4] = xi1; e[5] = xi2;
}

// --------------------------------------------------------------------------
// 0) zero the stat accumulators
// --------------------------------------------------------------------------
__global__ void init_acc_kernel(float* __restrict__ acc) {
  for (int t = threadIdx.x; t < ACC_ZERO_N; t += blockDim.x) acc[t] = 0.0f;
}

// --------------------------------------------------------------------------
// 1) KNN: per-thread point, LDS-tiled candidates, top-40 insertion list
//    pd = 2*<xi,xj> - |xi|^2 - |xj|^2  (top_k of pd == nearest)
// --------------------------------------------------------------------------
__global__ void knn_kernel(const float* __restrict__ x, int* __restrict__ idx) {
  __shared__ float xs0[256], xs1[256], xs2[256], sqs[256];
  const int tid = threadIdx.x;
  const int g = blockIdx.x * 256 + tid;      // global point
  const int b = g >> 12;
  const int i = g & (N_ - 1);
  const float* xb = x + b * 3 * N_;
  const float xi0 = xb[i], xi1 = xb[N_ + i], xi2 = xb[2 * N_ + i];
  const float sqi = xi0 * xi0 + xi1 * xi1 + xi2 * xi2;

  float bd[K_];
  int   bi[K_];
#pragma unroll
  for (int r = 0; r < K_; ++r) { bd[r] = -1e30f; bi[r] = 0; }

  for (int jt = 0; jt < N_; jt += 256) {
    int j = jt + tid;
    float a0 = xb[j], a1 = xb[N_ + j], a2 = xb[2 * N_ + j];
    xs0[tid] = a0; xs1[tid] = a1; xs2[tid] = a2;
    sqs[tid] = a0 * a0 + a1 * a1 + a2 * a2;
    __syncthreads();
    for (int t = 0; t < 256; ++t) {
      float pd = 2.0f * (xi0 * xs0[t] + xi1 * xs1[t] + xi2 * xs2[t]) - sqi - sqs[t];
      if (pd > bd[K_ - 1]) {
        int p = K_ - 1;
        while (p > 0 && bd[p - 1] < pd) { bd[p] = bd[p - 1]; bi[p] = bi[p - 1]; --p; }
        bd[p] = pd; bi[p] = jt + t;
      }
    }
    __syncthreads();
  }
#pragma unroll
  for (int r = 0; r < K_; ++r) idx[g * K_ + r] = bi[r];
}

// --------------------------------------------------------------------------
// 2) Layer-1 BN stats via edge moments: m1[6], M2[21] (symmetric)
// --------------------------------------------------------------------------
__global__ void stats1_kernel(const float* __restrict__ x,
                              const int* __restrict__ idx,
                              float* __restrict__ acc) {
  float loc[27];
#pragma unroll
  for (int t = 0; t < 27; ++t) loc[t] = 0.0f;

  const int stride = gridDim.x * blockDim.x;
  for (int e = blockIdx.x * blockDim.x + threadIdx.x; e < CNT_; e += stride) {
    int b = e / (N_ * K_);
    int r = e - b * (N_ * K_);
    int i = r / K_;
    int j = idx[e];
    float ed[6];
    load_edge(x, b, i, j, ed);
#pragma unroll
    for (int c = 0; c < 6; ++c) {
      loc[c] += ed[c];
#pragma unroll
      for (int d = 0; d < 6; ++d)
        if (d >= c) loc[6 + sym6(c, d)] += ed[c] * ed[d];
    }
  }

  __shared__ float red[27];
  if (threadIdx.x < 27) red[threadIdx.x] = 0.0f;
  __syncthreads();
#pragma unroll
  for (int t = 0; t < 27; ++t) atomicAdd(&red[t], loc[t]);
  __syncthreads();
  if (threadIdx.x < 27) atomicAdd(&acc[threadIdx.x], red[threadIdx.x]);
}

// --------------------------------------------------------------------------
// 3) fold moments through W1 -> scale1/shift1
// --------------------------------------------------------------------------
__global__ void finalize1_kernel(const float* __restrict__ W1,
                                 const float* __restrict__ g1,
                                 const float* __restrict__ b1,
                                 float* __restrict__ acc) {
  int f = threadIdx.x;
  if (f >= F_) return;
  float w[6];
#pragma unroll
  for (int c = 0; c < 6; ++c) w[c] = W1[f * 6 + c];
  const float inv = 1.0f / (float)CNT_;
  float mean = 0.0f, ex2 = 0.0f;
#pragma unroll
  for (int c = 0; c < 6; ++c) {
    mean += w[c] * acc[OFF_M1 + c];
#pragma unroll
    for (int d = 0; d < 6; ++d)
      ex2 += w[c] * w[d] * acc[OFF_M2 + sym6(c, d)];
  }
  mean *= inv; ex2 *= inv;
  float var = ex2 - mean * mean;
  float sc = g1[f] * rsqrtf(var + EPS_);
  acc[OFF_SC1 + f] = sc;
  acc[OFF_SH1 + f] = b1[f] - mean * sc;
}

// --------------------------------------------------------------------------
// 4) Layer-2 BN stats: recompute a1 (8 points/WG), WMMA SYRK S += a1*a1^T,
//    plus first moment m1a.  128 threads = 4 waves; 320 cols = 10 K-steps.
// --------------------------------------------------------------------------
__global__ void
__launch_bounds__(128)
stats2_kernel(const float* __restrict__ x, const int* __restrict__ idx,
              const float* __restrict__ W1, float* __restrict__ acc) {
  __shared__ __bf16 A1t[F_ * 328];   // [f][col], col contiguous, padded stride
  __shared__ float  W1s[F_ * 6];
  __shared__ float  sc1[F_], sh1[F_];

  const int tid = threadIdx.x;
  const int pbase = blockIdx.x * 8;
  const int b = pbase >> 12;
  const int n0 = pbase & (N_ - 1);

  for (int t = tid; t < F_ * 6; t += 128) W1s[t] = W1[t];
  if (tid < F_) { sc1[tid] = acc[OFF_SC1 + tid]; sh1[tid] = acc[OFF_SH1 + tid]; }
  __syncthreads();

  // build a1 (post BN1+LReLU) columns, bf16
  for (int c = tid; c < 320; c += 128) {
    int p = c / K_, k = c - p * K_;
    int n = n0 + p;
    int j = idx[(b * N_ + n) * K_ + k];
    float ed[6];
    load_edge(x, b, n, j, ed);
#pragma unroll
    for (int f = 0; f < F_; ++f) {
      const float* wf = &W1s[f * 6];
      float y = wf[0]*ed[0] + wf[1]*ed[1] + wf[2]*ed[2] +
                wf[3]*ed[3] + wf[4]*ed[4] + wf[5]*ed[5];
      float a = y * sc1[f] + sh1[f];
      a = (a >= 0.0f) ? a : SLOPE_ * a;
      A1t[f * 328 + c] = (__bf16)a;
    }
  }
  __syncthreads();

  // SYRK: S(64x64) += A1(64x320) * A1^T.  wave w owns m-tile w, 4 n-tiles.
  const int wave = tid >> 5, lane = tid & 31;
  const int ln = lane & 15;
  const bool lo = lane < 16;
  v8f accv[4];
#pragma unroll
  for (int nt = 0; nt < 4; ++nt) accv[nt] = (v8f){};

  for (int ks = 0; ks < 10; ++ks) {
    const int kc = ks * 32;
    v16bf afr;
    const int m = wave * 16 + ln;
#pragma unroll
    for (int v = 0; v < 8; ++v) {
      int kp = kc + ((v < 4) ? 2 * v : 2 * v + 8) + (lo ? 0 : 8);
      afr[2 * v]     = A1t[m * 328 + kp];
      afr[2 * v + 1] = A1t[m * 328 + kp + 1];
    }
#pragma unroll
    for (int nt = 0; nt < 4; ++nt) {
      v16bf bfr;
      const int n = nt * 16 + ln;
#pragma unroll
      for (int v = 0; v < 8; ++v) {
        int kp = kc + ((v < 4) ? 2 * v : 2 * v + 8) + (lo ? 0 : 8);
        bfr[2 * v]     = A1t[n * 328 + kp];
        bfr[2 * v + 1] = A1t[n * 328 + kp + 1];
      }
      accv[nt] = __builtin_amdgcn_wmma_f32_16x16x32_bf16(
          false, afr, false, bfr, (short)0, accv[nt], false, false);
    }
  }
#pragma unroll
  for (int nt = 0; nt < 4; ++nt)
#pragma unroll
    for (int r = 0; r < 8; ++r) {
      int mrow = wave * 16 + r + (lo ? 0 : 8);
      int ncol = nt * 16 + ln;
      atomicAdd(&acc[OFF_S + mrow * F_ + ncol], accv[nt][r]);
    }

  // first moment of a1
  __syncthreads();
  if (tid < F_) {
    float s = 0.0f;
    for (int c = 0; c < 320; ++c) s += (float)A1t[tid * 328 + c];
    atomicAdd(&acc[OFF_M1A + tid], s);
  }
}

// --------------------------------------------------------------------------
// 5) fold a1 moments through W2 -> scale2/shift2
// --------------------------------------------------------------------------
__global__ void finalize2_kernel(const float* __restrict__ W2,
                                 const float* __restrict__ g2,
                                 const float* __restrict__ b2,
                                 float* __restrict__ acc) {
  int f = threadIdx.x;
  if (f >= F_) return;
  const float inv = 1.0f / (float)CNT_;
  float mean = 0.0f, ex2 = 0.0f;
  for (int g = 0; g < F_; ++g) {
    float wg = W2[f * F_ + g];
    mean += wg * acc[OFF_M1A + g];
    float inner = 0.0f;
    for (int h = 0; h < F_; ++h) inner += W2[f * F_ + h] * acc[OFF_S + g * F_ + h];
    ex2 += wg * inner;
  }
  mean *= inv; ex2 *= inv;
  float var = ex2 - mean * mean;
  float sc = g2[f] * rsqrtf(var + EPS_);
  acc[OFF_SC2 + f] = sc;
  acc[OFF_SH2 + f] = b2[f] - mean * sc;
}

// --------------------------------------------------------------------------
// 6) main fused kernel: a1 -> WMMA GEMM with W2 -> BN2+LReLU -> max over K
//    128 threads = 4 waves; wave w owns point w (48 padded cols).
// --------------------------------------------------------------------------
__global__ void
__launch_bounds__(128)
conv2_kernel(const float* __restrict__ x, const int* __restrict__ idx,
             const float* __restrict__ W1, const float* __restrict__ W2,
             const float* __restrict__ acc, float* __restrict__ out) {
  __shared__ __bf16 W2s[F_ * F_];        // [m][k]
  __shared__ __bf16 A1s[192 * 66];       // [col][f], padded stride 66
  __shared__ float  stage[4][16 * 17];   // per-wave [n][m] tile staging
  __shared__ float  W1s[F_ * 6];
  __shared__ float  sc1[F_], sh1[F_], sc2[F_], sh2[F_];

  const int tid = threadIdx.x;
  const int pbase = blockIdx.x * 4;
  const int b = pbase >> 12;
  const int n0 = pbase & (N_ - 1);

  for (int t = tid; t < F_ * F_; t += 128) W2s[t] = (__bf16)W2[t];
  for (int t = tid; t < F_ * 6; t += 128) W1s[t] = W1[t];
  if (tid < F_) {
    sc1[tid] = acc[OFF_SC1 + tid]; sh1[tid] = acc[OFF_SH1 + tid];
    sc2[tid] = acc[OFF_SC2 + tid]; sh2[tid] = acc[OFF_SH2 + tid];
  }
  __syncthreads();

  // build a1 columns (k<40 real, 40..47 zero pad)
  for (int c = tid; c < 192; c += 128) {
    int p = c / 48, k = c - p * 48;
    if (k < K_) {
      int n = n0 + p;
      int j = idx[(b * N_ + n) * K_ + k];
      float ed[6];
      load_edge(x, b, n, j, ed);
#pragma unroll
      for (int f = 0; f < F_; ++f) {
        const float* wf = &W1s[f * 6];
        float y = wf[0]*ed[0] + wf[1]*ed[1] + wf[2]*ed[2] +
                  wf[3]*ed[3] + wf[4]*ed[4] + wf[5]*ed[5];
        float a = y * sc1[f] + sh1[f];
        a = (a >= 0.0f) ? a : SLOPE_ * a;
        A1s[c * 66 + f] = (__bf16)a;
      }
    } else {
#pragma unroll
      for (int f = 0; f < F_; ++f) A1s[c * 66 + f] = (__bf16)0.0f;
    }
  }
  __syncthreads();

  const int wave = tid >> 5, lane = tid & 31;
  const int ln = lane & 15;
  const bool lo = lane < 16;
  const int colbase = wave * 48;

  for (int mt = 0; mt < 4; ++mt) {
    float rmax = -1e30f;
    for (int nt = 0; nt < 3; ++nt) {
      v8f accv = (v8f){};
#pragma unroll
      for (int ks = 0; ks < 2; ++ks) {
        v16bf afr, bfr;
        const int m = mt * 16 + ln;
#pragma unroll
        for (int v = 0; v < 8; ++v) {
          int kp = 32 * ks + ((v < 4) ? 2 * v : 2 * v + 8) + (lo ? 0 : 8);
          afr[2 * v]     = W2s[m * F_ + kp];
          afr[2 * v + 1] = W2s[m * F_ + kp + 1];
        }
        const int col = colbase + nt * 16 + ln;
#pragma unroll
        for (int v = 0; v < 8; ++v) {
          int kp = 32 * ks + ((v < 4) ? 2 * v : 2 * v + 8) + (lo ? 0 : 8);
          bfr[2 * v]     = A1s[col * 66 + kp];
          bfr[2 * v + 1] = A1s[col * 66 + kp + 1];
        }
        accv = __builtin_amdgcn_wmma_f32_16x16x32_bf16(
            false, afr, false, bfr, (short)0, accv, false, false);
      }
      // BN2 + LeakyReLU epilogue, stage tile [n][m]
#pragma unroll
      for (int r = 0; r < 8; ++r) {
        int m_loc = r + (lo ? 0 : 8);
        int m = mt * 16 + m_loc;
        float v = accv[r] * sc2[m] + sh2[m];
        v = (v >= 0.0f) ? v : SLOPE_ * v;
        stage[wave][ln * 17 + m_loc] = v;
      }
      __syncthreads();
      if (lo) {
#pragma unroll
        for (int n = 0; n < 16; ++n) {
          int gk = nt * 16 + n;
          if (gk < K_) rmax = fmaxf(rmax, stage[wave][n * 17 + ln]);
        }
      }
      __syncthreads();
    }
    if (lo) {
      int f = mt * 16 + ln;
      out[((size_t)b * F_ + f) * N_ + (n0 + wave)] = rmax;
    }
  }
}

// --------------------------------------------------------------------------
extern "C" void kernel_launch(void* const* d_in, const int* in_sizes, int n_in,
                              void* d_out, int out_size, void* d_ws, size_t ws_size,
                              hipStream_t stream) {
  const float* x  = (const float*)d_in[0];
  const float* W1 = (const float*)d_in[1];
  const float* g1 = (const float*)d_in[2];
  const float* b1 = (const float*)d_in[3];
  const float* W2 = (const float*)d_in[4];
  const float* g2 = (const float*)d_in[5];
  const float* b2 = (const float*)d_in[6];
  float* out = (float*)d_out;

  int*   idx = (int*)d_ws;
  float* acc = (float*)d_ws + IDX_ELEMS;

  init_acc_kernel<<<1, 256, 0, stream>>>(acc);
  knn_kernel<<<(B_ * N_) / 256, 256, 0, stream>>>(x, idx);
  stats1_kernel<<<512, 256, 0, stream>>>(x, idx, acc);
  finalize1_kernel<<<1, 64, 0, stream>>>(W1, g1, b1, acc);
  stats2_kernel<<<(B_ * N_) / 8, 128, 0, stream>>>(x, idx, W1, acc);
  finalize2_kernel<<<1, 64, 0, stream>>>(W2, g2, b2, acc);
  conv2_kernel<<<(B_ * N_) / 4, 128, 0, stream>>>(x, idx, W1, W2, acc, out);
}
